// DualDGCNN_57647051047340
// MI455X (gfx1250) — matrix operand order repaired and checked by reading, not compiled, not run on hardware
//
#include <hip/hip_runtime.h>

typedef __attribute__((ext_vector_type(16))) _Float16 v16h;
typedef __attribute__((ext_vector_type(8)))  _Float16 v8h;
typedef __attribute__((ext_vector_type(8)))  float    v8f;

namespace {
constexpr int kB = 32;      // batch
constexpr int kN = 1024;    // points
constexpr int kK = 20;      // neighbors
constexpr int kCat = 448;   // concat channels
constexpr float kSlope = 0.2f;
constexpr float kEps = 1e-5f;
}

struct h2pack { _Float16 x, y; };

// ---------------------------------------------------------------------------
// sq[b,n] = sum_c feat[b,c,n]^2
__global__ void sumsq_kernel(const float* __restrict__ feat, long long sb, int C,
                             float* __restrict__ sq) {
  int t = blockIdx.x * blockDim.x + threadIdx.x;
  if (t >= kB * kN) return;
  int b = t >> 10, n = t & 1023;
  const float* fb = feat + (long long)b * sb;
  float s = 0.f;
  for (int c = 0; c < C; ++c) { float v = fb[c * kN + n]; s += v * v; }
  sq[t] = s;
}

// ---------------------------------------------------------------------------
// top-k (k=20) of d[b,i,j] = 2*dot - sq_i - sq_j ; one thread per row.
__global__ void topk_kernel(const float* __restrict__ feat, long long sb, int C,
                            const float* __restrict__ sq, int* __restrict__ idx) {
  int t = blockIdx.x * blockDim.x + threadIdx.x;
  if (t >= kB * kN) return;
  int b = t >> 10, i = t & 1023;
  const float* fb = feat + (long long)b * sb;
  float xi[128];
  for (int c = 0; c < C; ++c) xi[c] = fb[c * kN + i];
  float sqi = sq[t];
  float bv[kK]; int bi[kK];
  for (int q = 0; q < kK; ++q) { bv[q] = -3.4e38f; bi[q] = 0; }
  const float* sqb = sq + (b << 10);
  for (int j = 0; j < kN; ++j) {
    float dot = 0.f;
    for (int c = 0; c < C; ++c) dot += xi[c] * fb[c * kN + j];
    float d = 2.f * dot - sqi - sqb[j];
    if (d > bv[kK - 1]) {
      int p = kK - 1;
      while (p > 0 && bv[p - 1] < d) { bv[p] = bv[p - 1]; bi[p] = bi[p - 1]; --p; }
      bv[p] = d; bi[p] = j;
    }
  }
  for (int q = 0; q < kK; ++q) idx[t * kK + q] = bi[q];
}

// ---------------------------------------------------------------------------
// Edge-conv BN stats pass: per-channel sum / sumsq of h = W.e over (b,n,k).
__global__ __launch_bounds__(32)
void econv_stats_kernel(const float* __restrict__ feat, long long sb, int C, int O,
                        const int* __restrict__ idx, const float* __restrict__ Wc,
                        float* __restrict__ sums) {
  __shared__ float e[32][257];
  int lane = threadIdx.x;
  int t = blockIdx.x * 32 + lane;          // tuple id, grid exact
  int kk = t % kK;
  int bn = t / kK;
  int b = bn >> 10, n = bn & 1023;
  const float* fb = feat + (long long)b * sb;
  int j = idx[bn * kK + kk];
  for (int c = 0; c < C; ++c) {
    float ctr = fb[c * kN + n];
    float nb  = fb[c * kN + j];
    e[lane][c]     = nb - ctr;
    e[lane][C + c] = ctr;
  }
  __syncthreads();
  int C2 = 2 * C;
  for (int o = 0; o < O; ++o) {
    const float* w = Wc + o * C2;
    float h = 0.f;
    for (int c = 0; c < C2; ++c) h += w[c] * e[lane][c];
    float h2 = h * h;
    for (int off = 16; off > 0; off >>= 1) {
      h  += __shfl_down(h,  off, 32);
      h2 += __shfl_down(h2, off, 32);
    }
    if (lane == 0) {
      atomicAdd(&sums[o], h);
      atomicAdd(&sums[O + o], h2);
    }
  }
}

__global__ void bn_finalize_kernel(const float* __restrict__ sums, int O, float cnt,
                                   const float* __restrict__ g, const float* __restrict__ beta,
                                   float* __restrict__ ss) {
  int o = blockIdx.x * blockDim.x + threadIdx.x;
  if (o >= O) return;
  float m = sums[o] / cnt;
  float v = sums[O + o] / cnt - m * m;
  float sc = g[o] * rsqrtf(v + kEps);
  ss[o] = sc;
  ss[O + o] = beta[o] - sc * m;
}

// ---------------------------------------------------------------------------
// Edge-conv apply: recompute h, BN+leaky, max over k -> outfeat[b, o, n]
__global__ __launch_bounds__(64)
void econv_apply_kernel(const float* __restrict__ feat, long long sb, int C, int O,
                        const int* __restrict__ idx, const float* __restrict__ Wc,
                        const float* __restrict__ ss,
                        float* __restrict__ outfeat, long long sbOut) {
  __shared__ float e[kK][257];
  __shared__ float ctr[128];
  __shared__ int jj[kK];
  int bn = blockIdx.x;
  int b = bn >> 10, n = bn & 1023;
  int tid = threadIdx.x;
  const float* fb = feat + (long long)b * sb;
  for (int c = tid; c < C; c += 64) ctr[c] = fb[c * kN + n];
  if (tid < kK) jj[tid] = idx[bn * kK + tid];
  __syncthreads();
  for (int q = tid; q < kK * C; q += 64) {
    int kk = q / C, c = q % C;
    float nb = fb[c * kN + jj[kk]];
    e[kk][c]     = nb - ctr[c];
    e[kk][C + c] = ctr[c];
  }
  __syncthreads();
  int C2 = 2 * C;
  for (int o = tid; o < O; o += 64) {
    const float* w = Wc + o * C2;
    float sc = ss[o], sh = ss[O + o];
    float mx = -3.4e38f;
    for (int kk = 0; kk < kK; ++kk) {
      float h = 0.f;
      for (int c = 0; c < C2; ++c) h += w[c] * e[kk][c];
      float y = sc * h + sh;
      y = (y >= 0.f) ? y : kSlope * y;
      mx = fmaxf(mx, y);
    }
    outfeat[(long long)b * sbOut + o * kN + n] = mx;
  }
}

// ---------------------------------------------------------------------------
__global__ void to_y_kernel(const float* __restrict__ feat, long long sb, int E,
                            float* __restrict__ y, int total) {
  int t = blockIdx.x * blockDim.x + threadIdx.x;
  if (t >= total) return;
  int e = t % E;
  int bn = t / E;
  int b = bn >> 10, n = bn & 1023;
  y[t] = feat[(long long)b * sb + e * kN + n];
}

__global__ void add_att_kernel(float* __restrict__ feat, long long sb, int E,
                               const float* __restrict__ att, int total) {
  int t = blockIdx.x * blockDim.x + threadIdx.x;
  if (t >= total) return;
  int e = t % E;
  int bn = t / E;
  int b = bn >> 10, n = bn & 1023;
  feat[(long long)b * sb + e * kN + n] += att[t];
}

// ---------------------------------------------------------------------------
// WMMA f16->f32 GEMM.  C[m,n] = sum_k A[m,k] * (transB ? B[n,k] : B[k,n]) + bias[n]
// 256 threads = 8 waves; block tile 64x64; wave = 16 rows x 32 cols (2 accums).
// LDS tiles staged in WMMA *fragment order* so each lane loads its whole
// fragment with two 16B ds_load_b128 reads. Requires M % 64 == 0, Kd % 32 == 0;
// N handled with clamp/predicate (EXEC stays all-1 for WMMA).
__global__ __launch_bounds__(256)
void gemm_wmma_kernel(const float* __restrict__ A, long long sAb, int lda,
                      const float* __restrict__ Bm, long long sBb, int ldb, int transB,
                      const float* __restrict__ bias,
                      float* __restrict__ C, long long sCb, int ldc,
                      int Kd, int Ncols) {
  // rows padded to 24 halves (48B): fragment rows stay 16B-aligned for b128.
  __shared__ _Float16 aF[4][32][24];   // [16-row group][lane][half slot]
  __shared__ _Float16 bF[4][32][24];   // [16-col group][lane][half slot]
  const int bz = blockIdx.z;
  const float* Ab = A + (long long)bz * sAb;
  const float* Bb = Bm + (long long)bz * sBb;
  float* Cb = C + (long long)bz * sCb;
  const int mBase = blockIdx.y * 64;
  const int nBase = blockIdx.x * 64;
  const int tid = threadIdx.x;
  const int wave = tid >> 5;           // 0..7
  const int lane = tid & 31;
  const int wr = wave >> 1;            // 0..3 row group
  const int wc = wave & 1;             // 0..1 col pair group
  v8f accL = {}, accR = {};
  for (int k0 = 0; k0 < Kd; k0 += 32) {
    // ---- stage A tile (64 x 32) as float2 pairs along k ----
#pragma unroll
    for (int q = 0; q < 4; ++q) {
      int u = tid + q * 256;           // 0..1023
      int m = u >> 4;                  // 0..63
      int kp = (u & 15) << 1;          // even k
      float2 fv = *(const float2*)&Ab[(long long)(mBase + m) * lda + (k0 + kp)];
      int g = (kp >> 3) & 1;           // lane group owning this k
      int hi = 8 * (kp >> 4) + (kp & 15) - 8 * g;   // half slot (even)
      h2pack hv = { (_Float16)fv.x, (_Float16)fv.y };
      *(h2pack*)&aF[m >> 4][16 * g + (m & 15)][hi] = hv;
    }
    // ---- stage B tile (32 x 64) ----
#pragma unroll
    for (int q = 0; q < 4; ++q) {
      int u = tid + q * 256;
      int n = u >> 4;                  // 0..63
      int kp = (u & 15) << 1;
      int nG = nBase + n;
      int nC = (nG < Ncols) ? nG : (Ncols - 1);     // clamp OOB reads
      float f0, f1;
      if (transB) {
        float2 fv = *(const float2*)&Bb[(long long)nC * ldb + (k0 + kp)];
        f0 = fv.x; f1 = fv.y;
      } else {
        f0 = Bb[(long long)(k0 + kp) * ldb + nC];
        f1 = Bb[(long long)(k0 + kp + 1) * ldb + nC];
      }
      h2pack hv = { (_Float16)f0, (_Float16)f1 };
      *(h2pack*)&bF[n >> 4][16 * ((kp >> 3) & 1) + (n & 15)][8 * (kp >> 4) + (kp & 7)] = hv;
    }
    __syncthreads();
    // prefetch next k-tile into cache while WMMAs run
    if (k0 + 32 < Kd) {
      __builtin_prefetch(&Ab[(long long)(mBase + (tid & 63)) * lda + (k0 + 32)], 0, 0);
      if (transB) {
        int nP = nBase + (tid & 63);
        nP = (nP < Ncols) ? nP : (Ncols - 1);
        __builtin_prefetch(&Bb[(long long)nP * ldb + (k0 + 32)], 0, 0);
      } else {
        __builtin_prefetch(&Bb[(long long)(k0 + 32 + (tid >> 3)) * ldb + nBase], 0, 0);
      }
    }
    // ---- fragment loads: two b128 per operand ----
    v8h a0 = *(const v8h*)&aF[wr][lane][0];
    v8h a1 = *(const v8h*)&aF[wr][lane][8];
    v16h a = __builtin_shufflevector(a0, a1, 0,1,2,3,4,5,6,7,8,9,10,11,12,13,14,15);
    v8h l0 = *(const v8h*)&bF[2 * wc][lane][0];
    v8h l1 = *(const v8h*)&bF[2 * wc][lane][8];
    v16h bL = __builtin_shufflevector(l0, l1, 0,1,2,3,4,5,6,7,8,9,10,11,12,13,14,15);
    v8h r0 = *(const v8h*)&bF[2 * wc + 1][lane][0];
    v8h r1 = *(const v8h*)&bF[2 * wc + 1][lane][8];
    v16h bR = __builtin_shufflevector(r0, r1, 0,1,2,3,4,5,6,7,8,9,10,11,12,13,14,15);
    accL = __builtin_amdgcn_wmma_f32_16x16x32_f16(false, a, false, bL, (short)0, accL,
                                                  false, false);
    accR = __builtin_amdgcn_wmma_f32_16x16x32_f16(false, a, false, bR, (short)0, accR,
                                                  false, false);
    __syncthreads();
  }
  const int g = lane >> 4;
  const int mRow = mBase + 16 * wr + 8 * g;     // acc VGPR r -> M = 8g + r
#pragma unroll
  for (int half = 0; half < 2; ++half) {
    int nGlob = nBase + 32 * wc + 16 * half + (lane & 15);
    if (nGlob < Ncols) {
      float bv = bias ? bias[nGlob] : 0.f;
      const v8f& acc = half ? accR : accL;
#pragma unroll
      for (int r = 0; r < 8; ++r)
        Cb[(long long)(mRow + r) * ldc + nGlob] = acc[r] + bv;
    }
  }
}

// ---------------------------------------------------------------------------
// MHA attention over batch axis L=32; one wave per (n, head).
__global__ __launch_bounds__(32)
void attention_kernel(const float* __restrict__ qkv, int E, float* __restrict__ out) {
  int n = blockIdx.x >> 2;
  int h = blockIdx.x & 3;
  int d = E >> 2;
  __shared__ float ks[32][33], vs[32][33];
  int i = threadIdx.x;                 // L index
  int E3 = 3 * E;
  const float* base = qkv + ((long long)i * kN + n) * E3 + h * d;
  float q[32];
  for (int dd = 0; dd < d; ++dd) {
    q[dd] = base[dd];
    ks[i][dd] = base[E + dd];
    vs[i][dd] = base[2 * E + dd];
  }
  __syncthreads();
  float s[32];
  float scale = rsqrtf((float)d);
  float mx = -3.4e38f;
  for (int j = 0; j < 32; ++j) {
    float dot = 0.f;
    for (int dd = 0; dd < d; ++dd) dot += q[dd] * ks[j][dd];
    s[j] = dot * scale;
    mx = fmaxf(mx, s[j]);
  }
  float sum = 0.f;
  for (int j = 0; j < 32; ++j) { s[j] = __expf(s[j] - mx); sum += s[j]; }
  float inv = 1.f / sum;
  for (int dd = 0; dd < d; ++dd) {
    float acc = 0.f;
    for (int j = 0; j < 32; ++j) acc += s[j] * vs[j][dd];
    out[((long long)i * kN + n) * E + h * d + dd] = acc * inv;
  }
}

// ---------------------------------------------------------------------------
__global__ __launch_bounds__(256)
void chan_stats_kernel(const float* __restrict__ x, int O, float* __restrict__ sums) {
  int o = blockIdx.x;
  float s = 0.f, s2 = 0.f;
  for (int t = threadIdx.x; t < kB * kN; t += 256) {
    int b = t >> 10, n = t & 1023;
    float v = x[((long long)b * O + o) * kN + n];
    s += v; s2 += v * v;
  }
  __shared__ float rs[256], rs2[256];
  rs[threadIdx.x] = s; rs2[threadIdx.x] = s2;
  __syncthreads();
  for (int off = 128; off > 0; off >>= 1) {
    if (threadIdx.x < off) {
      rs[threadIdx.x]  += rs[threadIdx.x + off];
      rs2[threadIdx.x] += rs2[threadIdx.x + off];
    }
    __syncthreads();
  }
  if (threadIdx.x == 0) { sums[o] = rs[0]; sums[O + o] = rs2[0]; }
}

__global__ __launch_bounds__(128)
void bn_leaky_pool_kernel(const float* __restrict__ x, const float* __restrict__ ss,
                          float* __restrict__ pooled) {
  int bo = blockIdx.x;                    // b*1024 + o
  int b = bo >> 10, o = bo & 1023;
  float sc = ss[o], sh = ss[1024 + o];
  const float* row = x + (long long)bo * kN;
  float mx = -3.4e38f, sm = 0.f;
  for (int n = threadIdx.x; n < kN; n += 128) {
    float y = sc * row[n] + sh;
    y = (y >= 0.f) ? y : kSlope * y;
    mx = fmaxf(mx, y); sm += y;
  }
  __shared__ float rm[128], rsum[128];
  rm[threadIdx.x] = mx; rsum[threadIdx.x] = sm;
  __syncthreads();
  for (int off = 64; off > 0; off >>= 1) {
    if (threadIdx.x < off) {
      rm[threadIdx.x] = fmaxf(rm[threadIdx.x], rm[threadIdx.x + off]);
      rsum[threadIdx.x] += rsum[threadIdx.x + off];
    }
    __syncthreads();
  }
  if (threadIdx.x == 0) {
    pooled[b * 2048 + o]        = rm[0];
    pooled[b * 2048 + 1024 + o] = rsum[0] / (float)kN;
  }
}

// ---------------------------------------------------------------------------
__global__ __launch_bounds__(32)
void lin_bn_leaky_kernel(const float* __restrict__ in, int Kd,
                         const float* __restrict__ W, const float* __restrict__ bias,
                         const float* __restrict__ g, const float* __restrict__ beta,
                         float* __restrict__ out, int O, float* __restrict__ out2) {
  int o = blockIdx.x;
  int b = threadIdx.x;                    // 0..31
  const float* w = W + (long long)o * Kd;
  const float* r = in + (long long)b * Kd;
  float v = bias ? bias[o] : 0.f;
  for (int k = 0; k < Kd; ++k) v += r[k] * w[k];
  float s = v, s2 = v * v;
  for (int off = 16; off > 0; off >>= 1) {
    s  += __shfl_xor(s,  off, 32);
    s2 += __shfl_xor(s2, off, 32);
  }
  float m = s / 32.f;
  float var = s2 / 32.f - m * m;
  float y = g[o] * (v - m) * rsqrtf(var + kEps) + beta[o];
  y = (y >= 0.f) ? y : kSlope * y;
  out[b * O + o] = y;
  if (out2) out2[b * O + o] = y;
}

__global__ void lin3_kernel(const float* __restrict__ in, const float* __restrict__ W,
                            const float* __restrict__ bias, float* __restrict__ out) {
  int t = blockIdx.x * blockDim.x + threadIdx.x;
  if (t >= 32 * 40) return;
  int b = t / 40, o = t % 40;
  float v = bias[o];
  const float* w = W + o * 256;
  const float* r = in + b * 256;
  for (int k = 0; k < 256; ++k) v += r[k] * w[k];
  out[t] = v;
}

// ---------------------------------------------------------------------------
extern "C" void kernel_launch(void* const* d_in, const int* in_sizes, int n_in,
                              void* d_out, int out_size, void* d_ws, size_t ws_size,
                              hipStream_t stream) {
  (void)in_sizes; (void)n_in; (void)out_size; (void)ws_size;
  const float* x      = (const float*)d_in[0];
  const float* bn5_b  = (const float*)d_in[1];
  const float* bn5_g  = (const float*)d_in[2];
  const float* bn6_b  = (const float*)d_in[3];
  const float* bn6_g  = (const float*)d_in[4];
  const float* bn7_b  = (const float*)d_in[5];
  const float* bn7_g  = (const float*)d_in[6];
  const float* conv5_w= (const float*)d_in[7];
  const float* lin1_w = (const float*)d_in[29];
  const float* lin2_b = (const float*)d_in[30];
  const float* lin2_w = (const float*)d_in[31];
  const float* lin3_b = (const float*)d_in[32];
  const float* lin3_w = (const float*)d_in[33];

  char* wp = (char*)d_ws;
  auto alloc = [&](size_t bytes) -> void* {
    void* p = wp; wp += (bytes + 255) & ~(size_t)255; return p;
  };
  float* hcat   = (float*)alloc((size_t)kB * kCat * kN * 4);     // 58.7 MB
  float* sq     = (float*)alloc((size_t)kB * kN * 4);
  int*   idxb   = (int*)  alloc((size_t)kB * kN * kK * 4);
  float* stats  = (float*)alloc(4096 * 4);
  float* ssbuf  = (float*)alloc(4096 * 4);
  float* big    = (float*)alloc((size_t)kB * 1024 * kN * 4);     // 134 MB, reused
  float* pooled = (float*)alloc((size_t)kB * 2048 * 4);
  float* f1     = (float*)alloc((size_t)kB * 512 * 4);
  float* f2     = (float*)alloc((size_t)kB * 256 * 4);
  float* ybuf    = big;
  float* qkvbuf  = big + 4194304;
  float* attout  = big + 16777216;
  float* attproj = big + 20971520;
  float* conv5out= big;                    // reused after MHA phase

  const int Cs[3] = {3, 32, 64};
  const int Os[3] = {32, 64, 128};
  const int offs[2][3] = {{0, 32, 96}, {224, 256, 320}};
  const long long sbCat = (long long)kCat * kN;

  float* dout = (float*)d_out;             // logits | f256 | f512

  for (int side = 0; side < 2; ++side) {
    int pbase = (side == 0) ? 8 : 34;
    for (int bi = 0; bi < 3; ++bi) {
      const float* bn_b   = (const float*)d_in[pbase + 7 * bi + 0];
      const float* bn_g   = (const float*)d_in[pbase + 7 * bi + 1];
      const float* conv_w = (const float*)d_in[pbase + 7 * bi + 2];
      const float* in_b   = (const float*)d_in[pbase + 7 * bi + 3];
      const float* in_w   = (const float*)d_in[pbase + 7 * bi + 4];
      const float* out_b  = (const float*)d_in[pbase + 7 * bi + 5];
      const float* out_w  = (const float*)d_in[pbase + 7 * bi + 6];
      int C = Cs[bi], O = Os[bi];
      const float* inFeat; long long inSb;
      if (bi == 0) { inFeat = x; inSb = 3 * kN; }
      else { inFeat = hcat + (long long)offs[side][bi - 1] * kN; inSb = sbCat; }
      float* outFeat = hcat + (long long)offs[side][bi] * kN;

      // kNN graph
      sumsq_kernel<<<(kB * kN + 255) / 256, 256, 0, stream>>>(inFeat, inSb, C, sq);
      topk_kernel<<<(kB * kN + 127) / 128, 128, 0, stream>>>(inFeat, inSb, C, sq, idxb);

      // edge conv + BN(train) + leaky + max_k
      hipMemsetAsync(stats, 0, 2 * O * sizeof(float), stream);
      econv_stats_kernel<<<(kB * kN * kK) / 32, 32, 0, stream>>>(
          inFeat, inSb, C, O, idxb, conv_w, stats);
      bn_finalize_kernel<<<(O + 63) / 64, 64, 0, stream>>>(
          stats, O, (float)(kB * kN * kK), bn_g, bn_b, ssbuf);
      econv_apply_kernel<<<kB * kN, 64, 0, stream>>>(
          inFeat, inSb, C, O, idxb, conv_w, ssbuf, outFeat, sbCat);

      // MHA over batch axis
      int total = kB * kN * O;
      to_y_kernel<<<(total + 255) / 256, 256, 0, stream>>>(outFeat, sbCat, O, ybuf, total);
      {
        dim3 grid((3 * O + 63) / 64, (kB * kN) / 64, 1);
        gemm_wmma_kernel<<<grid, 256, 0, stream>>>(
            ybuf, 0, O, in_w, 0, O, 1, in_b, qkvbuf, 0, 3 * O, O, 3 * O);
      }
      attention_kernel<<<kN * 4, 32, 0, stream>>>(qkvbuf, O, attout);
      {
        dim3 grid((O + 63) / 64, (kB * kN) / 64, 1);
        gemm_wmma_kernel<<<grid, 256, 0, stream>>>(
            attout, 0, O, out_w, 0, O, 1, out_b, attproj, 0, O, O, O);
      }
      add_att_kernel<<<(total + 255) / 256, 256, 0, stream>>>(
          outFeat, sbCat, O, attproj, total);
    }
  }

  // conv5: per-batch (1024 x 448) @ (448 x 1024) on WMMA
  {
    dim3 grid(kN / 64, 1024 / 64, kB);
    gemm_wmma_kernel<<<grid, 256, 0, stream>>>(
        conv5_w, 0, kCat, hcat, sbCat, kN, 0, nullptr,
        conv5out, (long long)1024 * kN, kN, kCat, kN);
  }
  chan_stats_kernel<<<1024, 256, 0, stream>>>(conv5out, 1024, stats);
  bn_finalize_kernel<<<(1024 + 63) / 64, 64, 0, stream>>>(
      stats, 1024, (float)(kB * kN), bn5_g, bn5_b, ssbuf);
  bn_leaky_pool_kernel<<<kB * 1024, 128, 0, stream>>>(conv5out, ssbuf, pooled);

  // head MLP (BN over batch of 32 -> one wave per channel)
  lin_bn_leaky_kernel<<<512, 32, 0, stream>>>(
      pooled, 2048, lin1_w, nullptr, bn6_g, bn6_b, f1, 512, dout + 1280 + 8192);
  lin_bn_leaky_kernel<<<256, 32, 0, stream>>>(
      f1, 512, lin2_w, lin2_b, bn7_g, bn7_b, f2, 256, dout + 1280);
  lin3_kernel<<<(32 * 40 + 127) / 128, 128, 0, stream>>>(f2, lin3_w, lin3_b, dout);
}
